// CircumpunctHead_64828236366562
// MI455X (gfx1250) — compile-verified, matrix operand order checked by reading, not run on hardware
//
#include <hip/hip_runtime.h>
#include <hip/hip_bf16.h>
#include <math.h>

// ---------------- problem constants ----------------
constexpr int B_  = 8;
constexpr int T_  = 2048;
constexpr int D_  = 128;
constexpr int BT_ = B_ * T_;
constexpr int NV_ = 64;
constexpr int DV_ = 32;

typedef __attribute__((ext_vector_type(16))) _Float16 v16h;
typedef __attribute__((ext_vector_type(8)))  float    v8f;

// ---------------- WMMA helper (CDNA5 f16 16x16x32, f32 accum) ----------------
__device__ __forceinline__ v8f wmma_f16(v16h a, v16h b, v8f c) {
  return __builtin_amdgcn_wmma_f32_16x16x32_f16(false, a, false, b, (short)0, c, false, false);
}

// A tile 16x32 (MxK) from row-major f16 [rows x ld]; also serves as B when the
// "B" matrix is a transposed row-major matrix (weight [N x K], K-matrix [S x D],
// or the transposed V matrix [D x T]).  Per lane: 2 x 16-byte contiguous loads.
__device__ __forceinline__ v16h tile_a_f16(const _Float16* base, int ld, int m0, int k0) {
  int lane = threadIdx.x & 31;
  int m  = lane & 15;
  int kb = (lane >> 4) << 3;          // lanes 16..31 hold K offset +8
  const _Float16* p = base + (size_t)(m0 + m) * ld + k0 + kb;
  v16h r;
#pragma unroll
  for (int i = 0; i < 4; ++i) {
    r[2*i]     = p[2*i];
    r[2*i+1]   = p[2*i+1];
    r[8+2*i]   = p[16+2*i];
    r[8+2*i+1] = p[16+2*i+1];
  }
  return r;
}

// Same A-tile pattern, fp32 source with convert-on-load.
__device__ __forceinline__ v16h tile_a_f32cvt(const float* base, int ld, int m0, int k0) {
  int lane = threadIdx.x & 31;
  int m  = lane & 15;
  int kb = (lane >> 4) << 3;
  const float* p = base + (size_t)(m0 + m) * ld + k0 + kb;
  v16h r;
#pragma unroll
  for (int i = 0; i < 4; ++i) {
    r[2*i]     = (_Float16)p[2*i];
    r[2*i+1]   = (_Float16)p[2*i+1];
    r[8+2*i]   = (_Float16)p[16+2*i];
    r[8+2*i+1] = (_Float16)p[16+2*i+1];
  }
  return r;
}

// A-tile built as exp(score - rowmax) for the attn@V pass (softmax recompute).
__device__ __forceinline__ v16h tile_a_exp(const float* S, int ld, int m0, int k0,
                                           const float* rowmax) {
  int lane = threadIdx.x & 31;
  int m  = lane & 15;
  int kb = (lane >> 4) << 3;
  const float* p = S + (size_t)(m0 + m) * ld + k0 + kb;
  float mx = rowmax[m0 + m];
  v16h r;
#pragma unroll
  for (int i = 0; i < 4; ++i) {
    r[2*i]     = (_Float16)__expf(p[2*i]      - mx);
    r[2*i+1]   = (_Float16)__expf(p[2*i+1]    - mx);
    r[8+2*i]   = (_Float16)__expf(p[16+2*i]   - mx);
    r[8+2*i+1] = (_Float16)__expf(p[16+2*i+1] - mx);
  }
  return r;
}

// ---------------- C-tile store variants ----------------
__device__ __forceinline__ void store_c_f16_gated(_Float16* dst, int ld, int m0, int n0,
                                                  v8f c, const float* gate) {
  int lane = threadIdx.x & 31;
  int n  = lane & 15;
  int mb = (lane >> 4) << 3;
#pragma unroll
  for (int r = 0; r < 8; ++r) {
    int m = m0 + mb + r;
    float s = gate ? gate[m] : 1.0f;
    dst[(size_t)m * ld + n0 + n] = (_Float16)(c[r] * s);
  }
}

// V stored transposed: vt laid out [B][D][T].
__device__ __forceinline__ void store_c_f16_vt(_Float16* vt, int m0, int n0, v8f c) {
  int lane = threadIdx.x & 31;
  int n  = lane & 15;
  int mb = (lane >> 4) << 3;
#pragma unroll
  for (int r = 0; r < 8; ++r) {
    int m = m0 + mb + r;                 // global row
    int b = m >> 11;                     // / T_
    int t = m & (T_ - 1);
    vt[((size_t)b * D_ + n0 + n) * T_ + t] = (_Float16)c[r];
  }
}

__device__ __forceinline__ void store_c_f32_bias(float* dst, int ld, int m0, int n0, v8f c,
                                                 const float* bias) {
  int lane = threadIdx.x & 31;
  int n  = lane & 15;
  int mb = (lane >> 4) << 3;
#pragma unroll
  for (int r = 0; r < 8; ++r) {
    int m = m0 + mb + r;
    dst[(size_t)m * ld + n0 + n] = c[r] + bias[n0 + n];
  }
}

__device__ __forceinline__ void store_gelu_f16(_Float16* dst, int ld, int m0, int n0, v8f c,
                                               const float* bias) {
  int lane = threadIdx.x & 31;
  int n  = lane & 15;
  int mb = (lane >> 4) << 3;
#pragma unroll
  for (int r = 0; r < 8; ++r) {
    int m = m0 + mb + r;
    int col = n0 + n;
    float xv = c[r] + bias[col];
    float g  = 0.5f * xv * (1.0f + erff(xv * 0.70710678118654752f));
    dst[(size_t)m * ld + col] = (_Float16)g;
  }
}

__device__ __forceinline__ void store_down(float* dst, int ld, int m0, int n0, v8f c,
                                           const float* bias, const float* resid) {
  int lane = threadIdx.x & 31;
  int n  = lane & 15;
  int mb = (lane >> 4) << 3;
#pragma unroll
  for (int r = 0; r < 8; ++r) {
    int m = m0 + mb + r;
    int col = n0 + n;
    size_t idx = (size_t)m * ld + col;
    dst[idx] = c[r] + bias[col] + resid[idx];
  }
}

// ---------------- small utility kernels ----------------
__global__ void f32_to_f16_kernel(const float* src, _Float16* dst, int n) {
  int i = blockIdx.x * blockDim.x + threadIdx.x;
  if (i < n) dst[i] = (_Float16)src[i];
}

__global__ void zero_kernel(float* p) { if (threadIdx.x == 0 && blockIdx.x == 0) *p = 0.0f; }

// LayerNorm over 128 cols, wave-per-row, fp32 src -> f16 dst.
__global__ void ln_f16_kernel(const float* __restrict__ x, const float* __restrict__ g,
                              const float* __restrict__ b, _Float16* __restrict__ out) {
  int wid  = threadIdx.x >> 5, lane = threadIdx.x & 31;
  int row  = blockIdx.x * (blockDim.x >> 5) + wid;
  const float* px = x + (size_t)row * D_;
  float v[4], s = 0.f, s2 = 0.f;
#pragma unroll
  for (int i = 0; i < 4; ++i) {
    float t = px[lane + 32 * i];
    v[i] = t; s += t; s2 += t * t;
  }
#pragma unroll
  for (int m = 16; m; m >>= 1) { s += __shfl_xor(s, m, 32); s2 += __shfl_xor(s2, m, 32); }
  float mean = s * (1.0f / D_);
  float var  = s2 * (1.0f / D_) - mean * mean;
  float inv  = rsqrtf(var + 1e-5f);
#pragma unroll
  for (int i = 0; i < 4; ++i) {
    int c = lane + 32 * i;
    out[(size_t)row * D_ + c] = (_Float16)((v[i] - mean) * inv * g[c] + b[c]);
  }
}

// ---------------- QKV projection GEMM ----------------
// One projection: acc = a(16x128) @ W^T tile-by-tile.  W is [out,in] row-major,
// so B[k][n] = W[n][k] -> A-pattern load.  Kept as a forced-inline helper with
// direct kernel-arg pointers so address-space inference yields global loads.
__device__ __forceinline__ void proj_accum(const v16h a[4], const _Float16* __restrict__ W,
                                           v8f acc[8]) {
  v8f zero = {};
#pragma unroll
  for (int n = 0; n < 8; ++n) acc[n] = zero;
#pragma unroll
  for (int kc = 0; kc < 4; ++kc) {
#pragma unroll
    for (int n = 0; n < 8; ++n) {
      v16h bt = tile_a_f16(W, D_, n * 16, kc * 32);
      acc[n] = wmma_f16(a[kc], bt, acc[n]);
    }
  }
}

__global__ void qkv_kernel(const _Float16* __restrict__ hc, const _Float16* __restrict__ wq,
                           const _Float16* __restrict__ wk, const _Float16* __restrict__ wv,
                           const float* __restrict__ gate, _Float16* __restrict__ qf,
                           _Float16* __restrict__ kf, _Float16* __restrict__ vt) {
  int wid = threadIdx.x >> 5;
  int m0  = (blockIdx.x * 4 + wid) * 16;
  v16h a[4];
#pragma unroll
  for (int kc = 0; kc < 4; ++kc) a[kc] = tile_a_f16(hc, D_, m0, kc * 32);

  v8f acc[8];
  proj_accum(a, wq, acc);
#pragma unroll
  for (int n = 0; n < 8; ++n) store_c_f16_gated(qf, D_, m0, n * 16, acc[n], gate);

  proj_accum(a, wk, acc);
#pragma unroll
  for (int n = 0; n < 8; ++n) store_c_f16_gated(kf, D_, m0, n * 16, acc[n], gate);

  proj_accum(a, wv, acc);
#pragma unroll
  for (int n = 0; n < 8; ++n) store_c_f16_vt(vt, m0, n * 16, acc[n]);
}

// ---------------- scores = Q @ K^T / sqrt(D) + mask -> alignment ----------------
// Fused online softmax statistics (rowmax / rowsum), flash-style, so the stats
// pass never has to re-read the 134MB alignment tensor.
__global__ void scores_kernel(const _Float16* __restrict__ qf, const _Float16* __restrict__ kf,
                              const float* __restrict__ mask, float* __restrict__ align,
                              float* __restrict__ rowmax, float* __restrict__ rowsum) {
  int b   = blockIdx.y;
  int t0  = blockIdx.x * 16;
  int wid = threadIdx.x >> 5, lane = threadIdx.x & 31;
  const _Float16* Qb = qf + (size_t)b * T_ * D_;
  const _Float16* Kb = kf + (size_t)b * T_ * D_;
  float* Ab = align + (size_t)b * T_ * T_;

  v16h aq[4];
#pragma unroll
  for (int kc = 0; kc < 4; ++kc) aq[kc] = tile_a_f16(Qb, D_, t0, kc * 32);

  int nn = lane & 15, mb = (lane >> 4) << 3;
  float rm[8], rs[8];
#pragma unroll
  for (int r = 0; r < 8; ++r) { rm[r] = -3.4e38f; rs[r] = 0.f; }

  const float inv_scale = 0.08838834764831845f;  // 1/sqrt(128)
  for (int st = wid; st < T_ / 16; st += 4) {
    int s0 = st * 16;
    if (st + 4 < T_ / 16)
      __builtin_prefetch(Kb + (size_t)(s0 + 64) * D_, 0, 1);
    v8f acc = {};
#pragma unroll
    for (int kc = 0; kc < 4; ++kc) {
      v16h bt = tile_a_f16(Kb, D_, s0, kc * 32);  // K^T: B[k][s] = K[s][k]
      acc = wmma_f16(aq[kc], bt, acc);
    }
    // store scores tile + online per-row max/sum over the 16 lanes of each half
#pragma unroll
    for (int r = 0; r < 8; ++r) {
      int m = t0 + mb + r;
      size_t idx = (size_t)m * T_ + s0 + nn;
      float val = acc[r] * inv_scale + mask[idx];
      Ab[idx] = val;
      float tm = val;
#pragma unroll
      for (int xm = 1; xm < 16; xm <<= 1) tm = fmaxf(tm, __shfl_xor(tm, xm, 32));
      float nm = fmaxf(rm[r], tm);
      float e = __expf(val - nm);
#pragma unroll
      for (int xm = 1; xm < 16; xm <<= 1) e += __shfl_xor(e, xm, 32);
      rs[r] = rs[r] * __expf(rm[r] - nm) + e;
      rm[r] = nm;
    }
  }

  // cross-wave combine of per-row (m,s) partials
  __shared__ float smM[4][16], smS[4][16];
  if (nn == 0) {
#pragma unroll
    for (int r = 0; r < 8; ++r) { smM[wid][mb + r] = rm[r]; smS[wid][mb + r] = rs[r]; }
  }
  __syncthreads();
  if (threadIdx.x < 16) {
    int row = threadIdx.x;
    float M = smM[0][row];
#pragma unroll
    for (int w = 1; w < 4; ++w) M = fmaxf(M, smM[w][row]);
    float S = 0.f;
#pragma unroll
    for (int w = 0; w < 4; ++w) S += smS[w][row] * __expf(smM[w][row] - M);
    rowmax[(size_t)b * T_ + t0 + row] = M;
    rowsum[(size_t)b * T_ + t0 + row] = S;
  }
}

// ---------------- center_out = softmax(scores) @ V  (+x residual) ----------------
// Split-K across the 4 waves (each reads 1/4 of the score row once), LDS reduce.
__global__ void av_kernel(const float* __restrict__ align, const float* __restrict__ rowmax,
                          const float* __restrict__ rowsum, const _Float16* __restrict__ vt,
                          const float* __restrict__ x, float* __restrict__ center) {
  int b   = blockIdx.y;
  int t0  = blockIdx.x * 16;
  int wid = threadIdx.x >> 5, lane = threadIdx.x & 31;
  const float*    Sb  = align + (size_t)b * T_ * T_;
  const float*    rm  = rowmax + (size_t)b * T_;
  const _Float16* Vtb = vt + (size_t)b * D_ * T_;

  v8f zero = {};
  v8f acc[8];
#pragma unroll
  for (int n = 0; n < 8; ++n) acc[n] = zero;

  int kbeg = wid * (T_ / 4);
  int kend = kbeg + (T_ / 4);
  for (int k0 = kbeg; k0 < kend; k0 += 32) {
    if (k0 + 32 < kend)
      __builtin_prefetch(Sb + (size_t)t0 * T_ + k0 + 32, 0, 1);
    v16h a = tile_a_exp(Sb, T_, t0, k0, rm);
#pragma unroll
    for (int n = 0; n < 8; ++n) {
      v16h bt = tile_a_f16(Vtb, T_, n * 16, k0);  // B[k][n] = Vt[n][k] (contiguous)
      acc[n] = wmma_f16(a, bt, acc[n]);
    }
  }

  // cross-wave reduction in LDS
  __shared__ float red[4][16][D_];
  int nn = lane & 15, mb = (lane >> 4) << 3;
#pragma unroll
  for (int n = 0; n < 8; ++n)
#pragma unroll
    for (int r = 0; r < 8; ++r)
      red[wid][mb + r][n * 16 + nn] = acc[n][r];
  __syncthreads();

  for (int e = threadIdx.x; e < 16 * D_; e += 128) {
    int row = e >> 7, col = e & (D_ - 1);
    float s = red[0][row][col] + red[1][row][col] + red[2][row][col] + red[3][row][col];
    size_t gm = (size_t)b * T_ + t0 + row;
    size_t idx = gm * D_ + col;
    center[idx] = x[idx] + s / rowsum[gm];
  }
}

// ---------------- spec_proj = spectral_emb @ W_spec^T  [64,32] ----------------
__global__ void specproj_kernel(const float* __restrict__ spec, const float* __restrict__ wsp,
                                float* __restrict__ sp) {
  int i = blockIdx.x * blockDim.x + threadIdx.x;
  if (i >= NV_ * DV_) return;
  int r = i >> 5, c = i & 31;
  float s = 0.f;
#pragma unroll
  for (int k = 0; k < 16; ++k) s += spec[r * 16 + k] * wsp[c * 16 + k];
  sp[i] = s;
}

// ---------------- boundary stage 1: v_scores + global mean-abs ----------------
__global__ void boundary1_kernel(const float* __restrict__ x, const float* __restrict__ bn_g,
                                 const float* __restrict__ bn_b, const float* __restrict__ Wg_w,
                                 const float* __restrict__ Wg_b, const float* __restrict__ nb6,
                                 const float* __restrict__ g2v, const float* __restrict__ vemb,
                                 const float* __restrict__ sp, const float* __restrict__ adj,
                                 float* __restrict__ vsc, float* __restrict__ abssum) {
  int wid = threadIdx.x >> 5, lane = threadIdx.x & 31;
  int t = blockIdx.x * (blockDim.x >> 5) + wid;
  const float* px = x + (size_t)t * D_;

  // boundary LayerNorm
  float v[4], s = 0.f, s2 = 0.f;
#pragma unroll
  for (int i = 0; i < 4; ++i) {
    float tv = px[lane + 32 * i];
    v[i] = tv; s += tv; s2 += tv * tv;
  }
#pragma unroll
  for (int m = 16; m; m >>= 1) { s += __shfl_xor(s, m, 32); s2 += __shfl_xor(s2, m, 32); }
  float mean = s * (1.0f / D_), var = s2 * (1.0f / D_) - mean * mean;
  float inv = rsqrtf(var + 1e-5f);
  float hb[4];
#pragma unroll
  for (int i = 0; i < 4; ++i) {
    int c = lane + 32 * i;
    hb[i] = (v[i] - mean) * inv * bn_g[c] + bn_b[c];
  }

  // gates_6d = sigmoid(h_b @ Wg^T + Wg_b + node_bias)
  float gate6[6];
#pragma unroll
  for (int j = 0; j < 6; ++j) {
    float pj = 0.f;
#pragma unroll
    for (int i = 0; i < 4; ++i) pj += hb[i] * Wg_w[j * D_ + lane + 32 * i];
#pragma unroll
    for (int m = 16; m; m >>= 1) pj += __shfl_xor(pj, m, 32);
    gate6[j] = 1.0f / (1.0f + __expf(-(pj + Wg_b[j] + nb6[j])));
  }

  // gate_proj[lane] (lane = vertex-dim 0..31)
  float gp = 0.f;
#pragma unroll
  for (int j = 0; j < 6; ++j) gp += gate6[j] * g2v[lane * 6 + j];

  // v0[s] = gp . (vertex_emb[s] + 0.1*spec_proj[s]); lane handles s=lane, lane+32
  float v0a = 0.f, v0b = 0.f;
  for (int u = 0; u < 32; ++u) {
    float gpu = __shfl(gp, u, 32);
    v0a += gpu * (vemb[lane * DV_ + u]        + 0.1f * sp[lane * DV_ + u]);
    v0b += gpu * (vemb[(lane + 32) * DV_ + u] + 0.1f * sp[(lane + 32) * DV_ + u]);
  }
  // vs = v0 + 0.1 * v0 @ adj
  float vsa = v0a, vsb = v0b;
  for (int u = 0; u < 64; ++u) {
    float v0u = (u < 32) ? __shfl(v0a, u, 32) : __shfl(v0b, u - 32, 32);
    vsa += 0.1f * v0u * adj[u * NV_ + lane];
    vsb += 0.1f * v0u * adj[u * NV_ + lane + 32];
  }
  vsc[(size_t)t * NV_ + lane]      = vsa;
  vsc[(size_t)t * NV_ + lane + 32] = vsb;

  float ab = fabsf(vsa) + fabsf(vsb);
#pragma unroll
  for (int m = 16; m; m >>= 1) ab += __shfl_xor(ab, m, 32);
  if (lane == 0) atomicAdd(abssum, ab);
}

// ---------------- boundary stage 2: softmax, top-8 routing, vertex ctx ----------------
__global__ void boundary2_kernel(const float* __restrict__ vsc, const float* __restrict__ abssum,
                                 const float* __restrict__ vemb, const float* __restrict__ v2h,
                                 const float* __restrict__ x, float* __restrict__ pi_out,
                                 float* __restrict__ boundary) {
  int wid = threadIdx.x >> 5, lane = threadIdx.x & 31;
  int t = blockIdx.x * (blockDim.x >> 5) + wid;

  float mean_abs = fmaxf(abssum[0] * (1.0f / ((float)BT_ * NV_)), 1e-6f);
  float va = vsc[(size_t)t * NV_ + lane];
  float vb = vsc[(size_t)t * NV_ + lane + 32];
  float sa = va / mean_abs * 3.0f, sb = vb / mean_abs * 3.0f;

  // softmax over 64
  float mx = fmaxf(sa, sb);
#pragma unroll
  for (int m = 16; m; m >>= 1) mx = fmaxf(mx, __shfl_xor(mx, m, 32));
  float ea = __expf(sa - mx), eb = __expf(sb - mx);
  float sum = ea + eb;
#pragma unroll
  for (int m = 16; m; m >>= 1) sum += __shfl_xor(sum, m, 32);
  float pa = ea / sum, pb = eb / sum;  // soft_pi

  // top-8 by rank counting (stable by index for ties)
  int ra = 0, rb = 0;
  for (int u = 0; u < 64; ++u) {
    float pu = (u < 32) ? __shfl(pa, u, 32) : __shfl(pb, u - 32, 32);
    ra += (pu > pa) || (pu == pa && u < lane);
    rb += (pu > pb) || (pu == pb && u < lane + 32);
  }
  float ha  = (ra < 8) ? pa : 0.f;
  float hbv = (rb < 8) ? pb : 0.f;
  float hs = ha + hbv;
#pragma unroll
  for (int m = 16; m; m >>= 1) hs += __shfl_xor(hs, m, 32);
  float invn = 1.0f / (hs + 1e-8f);
  float pia = ha * invn, pib = hbv * invn;  // forward value of pi == hard_pi
  pi_out[(size_t)t * NV_ + lane]      = pia;
  pi_out[(size_t)t * NV_ + lane + 32] = pib;

  // vertex_ctx[v] = sum_s pi[s]*vemb[s][v], v = lane
  float vc = 0.f;
  for (int s = 0; s < 64; ++s) {
    float ps = (s < 32) ? __shfl(pia, s, 32) : __shfl(pib, s - 32, 32);
    vc += ps * vemb[s * DV_ + lane];
  }
  // boundary[t][d] = x[t][d] + sum_v vc[v]*v2h[d][v]
  float acc[4] = {0.f, 0.f, 0.f, 0.f};
  for (int u = 0; u < 32; ++u) {
    float vcv = __shfl(vc, u, 32);
#pragma unroll
    for (int i = 0; i < 4; ++i) acc[i] += vcv * v2h[(lane + 32 * i) * DV_ + u];
  }
#pragma unroll
  for (int i = 0; i < 4; ++i) {
    int c = lane + 32 * i;
    boundary[(size_t)t * D_ + c] = x[(size_t)t * D_ + c] + acc[i];
  }
}

// ---------------- phi_input = [center|boundary] @ pc_w^T + pc_b ----------------
__global__ void field_pc_kernel(const float* __restrict__ center,
                                const float* __restrict__ boundary,
                                const _Float16* __restrict__ pc16,
                                const float* __restrict__ pc_b, float* __restrict__ phi) {
  int wid = threadIdx.x >> 5;
  int m0  = (blockIdx.x * 4 + wid) * 16;
  v8f zero = {};
  v8f acc[8];
#pragma unroll
  for (int n = 0; n < 8; ++n) acc[n] = zero;
#pragma unroll
  for (int kc = 0; kc < 8; ++kc) {
    int k0 = kc * 32;
    const float* src = (k0 < D_) ? center : boundary;   // kc is unroll-constant
    v16h a = tile_a_f32cvt(src, D_, m0, k0 & (D_ - 1));
#pragma unroll
    for (int n = 0; n < 8; ++n) {
      v16h bt = tile_a_f16(pc16, 2 * D_, n * 16, k0);  // pc_w [128,256]
      acc[n] = wmma_f16(a, bt, acc[n]);
    }
  }
#pragma unroll
  for (int n = 0; n < 8; ++n) store_c_f32_bias(phi, D_, m0, n * 16, acc[n], pc_b);
}

// ---------------- FFN up: gelu(phn @ pu_w^T + pu_b) ----------------
__global__ void ffn_up_kernel(const _Float16* __restrict__ phn, const _Float16* __restrict__ pu16,
                              const float* __restrict__ pu_b, _Float16* __restrict__ uact) {
  int wid = threadIdx.x >> 5;
  int m0  = blockIdx.x * 16;
  v8f zero = {};
  v8f acc[4];
#pragma unroll
  for (int j = 0; j < 4; ++j) acc[j] = zero;
#pragma unroll
  for (int kc = 0; kc < 4; ++kc) {
    v16h a = tile_a_f16(phn, D_, m0, kc * 32);
#pragma unroll
    for (int j = 0; j < 4; ++j) {
      int n0 = (wid * 4 + j) * 16;
      v16h bt = tile_a_f16(pu16, D_, n0, kc * 32);  // pu_w [256,128]
      acc[j] = wmma_f16(a, bt, acc[j]);
    }
  }
#pragma unroll
  for (int j = 0; j < 4; ++j)
    store_gelu_f16(uact, 2 * D_, m0, (wid * 4 + j) * 16, acc[j], pu_b);
}

// ---------------- FFN down + residual -> field output ----------------
__global__ void ffn_down_kernel(const _Float16* __restrict__ uact,
                                const _Float16* __restrict__ pd16,
                                const float* __restrict__ pd_b, const float* __restrict__ phi,
                                float* __restrict__ outf) {
  int wid = threadIdx.x >> 5;
  int m0  = (blockIdx.x * 4 + wid) * 16;
  v8f zero = {};
  v8f acc[8];
#pragma unroll
  for (int n = 0; n < 8; ++n) acc[n] = zero;
#pragma unroll
  for (int kc = 0; kc < 8; ++kc) {
    v16h a = tile_a_f16(uact, 2 * D_, m0, kc * 32);
#pragma unroll
    for (int n = 0; n < 8; ++n) {
      v16h bt = tile_a_f16(pd16, 2 * D_, n * 16, kc * 32);  // pd_w [128,256]
      acc[n] = wmma_f16(a, bt, acc[n]);
    }
  }
#pragma unroll
  for (int n = 0; n < 8; ++n) store_down(outf, D_, m0, n * 16, acc[n], pd_b, phi);
}

// ---------------- launch ----------------
extern "C" void kernel_launch(void* const* d_in, const int* in_sizes, int n_in,
                              void* d_out, int out_size, void* d_ws, size_t ws_size,
                              hipStream_t stream) {
  const float* x    = (const float*)d_in[0];
  const float* gate = (const float*)d_in[1];
  const float* mask = (const float*)d_in[2];
  const float* nb6  = (const float*)d_in[3];
  const float* vemb = (const float*)d_in[4];
  const float* adj  = (const float*)d_in[5];
  const float* spec = (const float*)d_in[6];
  const float* wsp  = (const float*)d_in[7];
  const float* cn_g = (const float*)d_in[8];
  const float* cn_b = (const float*)d_in[9];
  const float* Wq   = (const float*)d_in[10];
  const float* Wk   = (const float*)d_in[11];
  const float* Wv   = (const float*)d_in[12];
  const float* bn_g = (const float*)d_in[13];
  const float* bn_b = (const float*)d_in[14];
  const float* Wg_w = (const float*)d_in[15];
  const float* Wg_b = (const float*)d_in[16];
  const float* g2v  = (const float*)d_in[17];
  const float* v2h  = (const float*)d_in[18];
  const float* pc_w = (const float*)d_in[19];
  const float* pc_b = (const float*)d_in[20];
  const float* pn_g = (const float*)d_in[21];
  const float* pn_b = (const float*)d_in[22];
  const float* pu_w = (const float*)d_in[23];
  const float* pu_b = (const float*)d_in[24];
  const float* pd_w = (const float*)d_in[25];
  const float* pd_b = (const float*)d_in[26];

  float* out       = (float*)d_out;
  float* out_field = out;
  float* out_pi    = out + (size_t)BT_ * D_;
  float* out_align = out_pi + (size_t)BT_ * NV_;

  // carve workspace
  char* w = (char*)d_ws;
  size_t off = 0;
  auto carve = [&](size_t bytes) -> void* {
    void* p = (void*)(w + off);
    off += (bytes + 255) & ~((size_t)255);
    return p;
  };
  _Float16* hc   = (_Float16*)carve((size_t)BT_ * D_ * 2);
  _Float16* qf   = (_Float16*)carve((size_t)BT_ * D_ * 2);
  _Float16* kf   = (_Float16*)carve((size_t)BT_ * D_ * 2);
  _Float16* vt   = (_Float16*)carve((size_t)BT_ * D_ * 2);  // transposed: [B][D][T]
  float* rowmax  = (float*)carve((size_t)BT_ * 4);
  float* rowsum  = (float*)carve((size_t)BT_ * 4);
  float* center  = (float*)carve((size_t)BT_ * D_ * 4);
  float* boundary= (float*)carve((size_t)BT_ * D_ * 4);
  float* vsc     = (float*)carve((size_t)BT_ * NV_ * 4);
  float* sp      = (float*)carve((size_t)NV_ * DV_ * 4);
  float* abssum  = (float*)carve(256);
  float* phi     = (float*)carve((size_t)BT_ * D_ * 4);
  _Float16* phn  = (_Float16*)carve((size_t)BT_ * D_ * 2);
  _Float16* uact = (_Float16*)carve((size_t)BT_ * 2 * D_ * 2);
  _Float16* wq16 = (_Float16*)carve((size_t)D_ * D_ * 2);
  _Float16* wk16 = (_Float16*)carve((size_t)D_ * D_ * 2);
  _Float16* wv16 = (_Float16*)carve((size_t)D_ * D_ * 2);
  _Float16* pc16 = (_Float16*)carve((size_t)D_ * 2 * D_ * 2);
  _Float16* pu16 = (_Float16*)carve((size_t)2 * D_ * D_ * 2);
  _Float16* pd16 = (_Float16*)carve((size_t)D_ * 2 * D_ * 2);

  auto cvt = [&](const float* s, _Float16* d, int n) {
    f32_to_f16_kernel<<<(n + 255) / 256, 256, 0, stream>>>(s, d, n);
  };
  cvt(Wq, wq16, D_ * D_);
  cvt(Wk, wk16, D_ * D_);
  cvt(Wv, wv16, D_ * D_);
  cvt(pc_w, pc16, D_ * 2 * D_);
  cvt(pu_w, pu16, 2 * D_ * D_);
  cvt(pd_w, pd16, D_ * 2 * D_);

  // center path
  ln_f16_kernel<<<BT_ / 8, 256, 0, stream>>>(x, cn_g, cn_b, hc);
  qkv_kernel<<<BT_ / 64, 128, 0, stream>>>(hc, wq16, wk16, wv16, gate, qf, kf, vt);
  scores_kernel<<<dim3(T_ / 16, B_), 128, 0, stream>>>(qf, kf, mask, out_align, rowmax, rowsum);
  av_kernel<<<dim3(T_ / 16, B_), 128, 0, stream>>>(out_align, rowmax, rowsum, vt, x, center);

  // boundary path
  specproj_kernel<<<(NV_ * DV_ + 255) / 256, 256, 0, stream>>>(spec, wsp, sp);
  zero_kernel<<<1, 1, 0, stream>>>(abssum);
  boundary1_kernel<<<BT_ / 8, 256, 0, stream>>>(x, bn_g, bn_b, Wg_w, Wg_b, nb6, g2v, vemb, sp,
                                                adj, vsc, abssum);
  boundary2_kernel<<<BT_ / 8, 256, 0, stream>>>(vsc, abssum, vemb, v2h, x, out_pi, boundary);

  // field path
  field_pc_kernel<<<BT_ / 64, 128, 0, stream>>>(center, boundary, pc16, pc_b, phi);
  ln_f16_kernel<<<BT_ / 8, 256, 0, stream>>>(phi, pn_g, pn_b, phn);
  ffn_up_kernel<<<BT_ / 16, 128, 0, stream>>>(phn, pu16, pu_b, uact);
  ffn_down_kernel<<<BT_ / 64, 128, 0, stream>>>(uact, pd16, pd_b, phi, out_field);

  (void)in_sizes; (void)n_in; (void)out_size; (void)ws_size;
}
